// SelfAttention_28157805593393
// MI455X (gfx1250) — compile-verified, matrix operand order checked by reading, not run on hardware
//
#include <hip/hip_runtime.h>
#include <hip/hip_bf16.h>

typedef __bf16 bf16_t;
typedef __attribute__((ext_vector_type(16))) __bf16 bf16x16;
typedef __attribute__((ext_vector_type(8)))  __bf16 bf16x8;
typedef __attribute__((ext_vector_type(8)))  float  f32x8;

#define DM   1024
#define SEQ  2048
#define NB   2
#define NH   16
#define HDIM 64
#define ROWS (NB * SEQ)   // 4096

// Load a 16x32 bf16 WMMA A-fragment (or 32x16 B-fragment, same lane layout)
// from a row-major source whose contraction dim is contiguous.
// Lane i=lane&15 owns row i; VGPRs 0..3 hold K = kbase+8*hh+0..7,
// VGPRs 4..7 hold K = kbase+16+8*hh+0..7  (hh = lane>>4). Two b128 loads.
__device__ __forceinline__ bf16x16 load_frag(const bf16_t* __restrict__ src,
                                             int stride, int kbase, int lane) {
  const int i  = lane & 15;
  const int hh = lane >> 4;
  const bf16_t* p = src + (size_t)i * stride + (kbase + 8 * hh);
  union { bf16x16 v; bf16x8 h[2]; } u;
  u.h[0] = *(const bf16x8*)(p);
  u.h[1] = *(const bf16x8*)(p + 16);
  return u.v;
}

__device__ __forceinline__ f32x8 wmma_bf16(bf16x16 a, bf16x16 b, f32x8 c) {
  return __builtin_amdgcn_wmma_f32_16x16x32_bf16(false, a, false, b,
                                                 (short)0, c, false, false);
}

__global__ void k_zero(float* out, int n) {
  int i = blockIdx.x * blockDim.x + threadIdx.x;
  if (i < n) out[i] = 0.f;
}

__global__ void k_cvt_x(const float* __restrict__ x, bf16_t* __restrict__ xb) {
  int i = blockIdx.x * blockDim.x + threadIdx.x;
  xb[i] = (bf16_t)x[i];
}

// Convert W [k][n] f32 -> Wt [n][k] bf16 (transposed, so WMMA B loads are
// contiguous along the contraction dim). blockIdx.y selects q/k/v.
__global__ void k_cvt_wt(const float* __restrict__ wq, const float* __restrict__ wk,
                         const float* __restrict__ wv, bf16_t* __restrict__ wt) {
  int idx = blockIdx.x * blockDim.x + threadIdx.x;   // idx = n*DM + k
  int z = blockIdx.y;
  const float* w = (z == 0) ? wq : (z == 1) ? wk : wv;
  int n = idx >> 10, k = idx & (DM - 1);
  wt[(size_t)z * DM * DM + idx] = (bf16_t)w[(size_t)k * DM + n];
}

// QKV projection GEMM: [4096 x 1024] = X[4096 x 1024] * W[1024 x 1024] + b.
// One wave per 32x32 output tile (2x2 register block): 4 WMMAs per 4 fragment
// loads per k-step -> 1:1 wmma:vmem ratio. 32 k-steps over K=1024.
// Q,K stored row-major bf16; V stored transposed [b][d][s] for the P*V pass.
__global__ void k_qkv_gemm(const bf16_t* __restrict__ xb, const bf16_t* __restrict__ wt,
                           const float* __restrict__ bq, const float* __restrict__ bk,
                           const float* __restrict__ bv,
                           bf16_t* __restrict__ qb, bf16_t* __restrict__ kb,
                           bf16_t* __restrict__ vt) {
  const int lane = threadIdx.x & 31;
  const int wave = threadIdx.x >> 5;
  const int tile = blockIdx.x * 8 + wave;           // 4096 32x32 tiles / matrix
  const int z = blockIdx.y;
  const int m0 = (tile >> 5) << 5;                  // 128 row-tiles of 32
  const int n0 = (tile & 31) << 5;                  // 32 col-tiles of 32
  const bf16_t* a_base = xb + (size_t)m0 * DM;
  const bf16_t* b_base = wt + (size_t)z * DM * DM + (size_t)n0 * DM;

  f32x8 acc[2][2];
#pragma unroll
  for (int tm = 0; tm < 2; ++tm)
#pragma unroll
    for (int tn = 0; tn < 2; ++tn)
      acc[tm][tn] = (f32x8){0.f, 0.f, 0.f, 0.f, 0.f, 0.f, 0.f, 0.f};

  for (int k0 = 0; k0 < DM; k0 += 32) {
    bf16x16 a0 = load_frag(a_base,            DM, k0, lane);
    bf16x16 a1 = load_frag(a_base + 16 * DM,  DM, k0, lane);
    bf16x16 b0 = load_frag(b_base,            DM, k0, lane);
    bf16x16 b1 = load_frag(b_base + 16 * DM,  DM, k0, lane);
    acc[0][0] = wmma_bf16(a0, b0, acc[0][0]);
    acc[0][1] = wmma_bf16(a0, b1, acc[0][1]);
    acc[1][0] = wmma_bf16(a1, b0, acc[1][0]);
    acc[1][1] = wmma_bf16(a1, b1, acc[1][1]);
  }

  const int i = lane & 15, hh = lane >> 4;
  const float* bias = (z == 0) ? bq : (z == 1) ? bk : bv;
#pragma unroll
  for (int tm = 0; tm < 2; ++tm)
#pragma unroll
    for (int tn = 0; tn < 2; ++tn) {
      const int n = n0 + tn * 16 + i;
      const float bn = bias[n];
#pragma unroll
      for (int r = 0; r < 8; ++r) {
        const int m = m0 + tm * 16 + r + 8 * hh;    // C-fragment row mapping
        const float v = acc[tm][tn][r] + bn;
        if (z == 0)      qb[(size_t)m * DM + n] = (bf16_t)v;
        else if (z == 1) kb[(size_t)m * DM + n] = (bf16_t)v;
        else {
          const int bb = m >> 11, s = m & (SEQ - 1);
          vt[((size_t)bb * DM + n) * SEQ + s] = (bf16_t)v;  // V^T [b][d][s]
        }
      }
    }
}

// Flash attention: one wave per (batch, head, 16-query tile).
// Streams keys in chunks of 32; online softmax with shfl_xor row reductions;
// P re-striped C-layout -> A-layout through a 1KB LDS tile.
// Prefetches the next K/V chunk (global_prefetch path) behind softmax VALU.
__global__ void k_attn(const bf16_t* __restrict__ qb, const bf16_t* __restrict__ kb,
                       const bf16_t* __restrict__ vt, float* __restrict__ out) {
  __shared__ bf16_t pt[16 * 32];
  const int lane = threadIdx.x;
  const int i = lane & 15, hh = lane >> 4;
  const int bid  = blockIdx.x;                      // 2*16*128 = 4096 blocks
  const int b    = bid >> 11;
  const int rem  = bid & 2047;
  const int head = rem >> 7;
  const int q0   = (rem & 127) << 4;

  const bf16_t* qsrc = qb + ((size_t)(b * SEQ + q0)) * DM + head * HDIM;
  const bf16x16 qa0 = load_frag(qsrc, DM, 0, lane);
  const bf16x16 qa1 = load_frag(qsrc, DM, 32, lane);

  const bf16_t* kbase_h = kb + (size_t)b * SEQ * DM + head * HDIM;
  const bf16_t* vbase_h = vt + ((size_t)b * DM + head * HDIM) * SEQ;

  float mrun[8], lrun[8];
  f32x8 ctx[4];
#pragma unroll
  for (int r = 0; r < 8; ++r) { mrun[r] = -1e30f; lrun[r] = 0.f; }
#pragma unroll
  for (int nt = 0; nt < 4; ++nt)
    ctx[nt] = (f32x8){0.f, 0.f, 0.f, 0.f, 0.f, 0.f, 0.f, 0.f};

  for (int kc = 0; kc < SEQ; kc += 32) {
    // ---- prefetch next chunk: 32 K rows + V stripes (global_prefetch) ----
    if (kc + 32 < SEQ) {
      __builtin_prefetch(kbase_h + (size_t)(kc + 32 + lane) * DM, 0, 1);
      __builtin_prefetch(vbase_h + (size_t)(lane * 2) * SEQ + (kc + 32), 0, 1);
    }

    // ---- scores for 32 keys: two 16x16 tiles, 2 wmma k-steps each ----
    f32x8 sf[2];
#pragma unroll
    for (int t = 0; t < 2; ++t) {
      const bf16_t* ksrc = kbase_h + (size_t)(kc + t * 16) * DM;
      bf16x16 kf0 = load_frag(ksrc, DM, 0, lane);
      bf16x16 kf1 = load_frag(ksrc, DM, 32, lane);
      f32x8 zacc = {0.f, 0.f, 0.f, 0.f, 0.f, 0.f, 0.f, 0.f};
      zacc = wmma_bf16(qa0, kf0, zacc);
      zacc = wmma_bf16(qa1, kf1, zacc);
#pragma unroll
      for (int r = 0; r < 8; ++r) sf[t][r] = zacc[r] * 0.125f;  // 1/sqrt(64)
    }

    // ---- online softmax: row max over 32 cols (one 16-lane half) ----
    float rmax[8];
#pragma unroll
    for (int r = 0; r < 8; ++r) {
      float v = fmaxf(sf[0][r], sf[1][r]);
      v = fmaxf(v, __shfl_xor(v, 1, 32));
      v = fmaxf(v, __shfl_xor(v, 2, 32));
      v = fmaxf(v, __shfl_xor(v, 4, 32));
      v = fmaxf(v, __shfl_xor(v, 8, 32));
      rmax[r] = v;
    }
    float alpha[8], rsum[8];
#pragma unroll
    for (int r = 0; r < 8; ++r) {
      float mnew = fmaxf(mrun[r], rmax[r]);
      alpha[r] = __expf(mrun[r] - mnew);
      mrun[r] = mnew;
      rsum[r] = 0.f;
    }
    // p = exp(s - m); write P tile (rows=query, cols=key-in-chunk) to LDS
#pragma unroll
    for (int t = 0; t < 2; ++t)
#pragma unroll
      for (int r = 0; r < 8; ++r) {
        float p = __expf(sf[t][r] - mrun[r]);
        rsum[r] += p;
        pt[(r + 8 * hh) * 32 + t * 16 + i] = (bf16_t)p;
      }
#pragma unroll
    for (int r = 0; r < 8; ++r) {
      float v = rsum[r];
      v += __shfl_xor(v, 1, 32);
      v += __shfl_xor(v, 2, 32);
      v += __shfl_xor(v, 4, 32);
      v += __shfl_xor(v, 8, 32);
      lrun[r] = lrun[r] * alpha[r] + v;
    }
#pragma unroll
    for (int nt = 0; nt < 4; ++nt)
#pragma unroll
      for (int r = 0; r < 8; ++r) ctx[nt][r] *= alpha[r];

    __syncthreads();
    // reload P as an A-fragment (lane=row layout) from LDS
    union { bf16x16 v; bf16x8 h[2]; } up;
    const bf16_t* prow = &pt[i * 32 + 8 * hh];
    up.h[0] = *(const bf16x8*)(prow);
    up.h[1] = *(const bf16x8*)(prow + 16);
    const bf16x16 pa = up.v;

    // ---- ctx += P[16x32] * V[32x64]: 4 wmma (head-dim col tiles) ----
#pragma unroll
    for (int nt = 0; nt < 4; ++nt) {
      const bf16_t* vsrc = vbase_h + (size_t)(nt * 16) * SEQ + kc;
      bf16x16 vb = load_frag(vsrc, SEQ, 0, lane);   // keys contiguous (V^T)
      ctx[nt] = wmma_bf16(pa, vb, ctx[nt]);
    }
    __syncthreads();
  }

  // ---- finalize: ctx/l, then mean over sequence -> atomic accumulate ----
  const float inv_s = 1.0f / (float)SEQ;
#pragma unroll
  for (int nt = 0; nt < 4; ++nt) {
    float v = 0.f;
#pragma unroll
    for (int r = 0; r < 8; ++r) v += ctx[nt][r] / lrun[r];
    const int d = head * HDIM + nt * 16 + i;
    atomicAdd(&out[b * DM + d], v * inv_s);
  }
}

extern "C" void kernel_launch(void* const* d_in, const int* in_sizes, int n_in,
                              void* d_out, int out_size, void* d_ws, size_t ws_size,
                              hipStream_t stream) {
  (void)in_sizes; (void)n_in; (void)out_size; (void)ws_size;
  const float* X  = (const float*)d_in[0];
  const float* Wq = (const float*)d_in[1];
  const float* bq = (const float*)d_in[2];
  const float* Wk = (const float*)d_in[3];
  const float* bk = (const float*)d_in[4];
  const float* Wv = (const float*)d_in[5];
  const float* bv = (const float*)d_in[6];
  float* out = (float*)d_out;

  char* ws = (char*)d_ws;
  const size_t MB = 1024 * 1024;
  bf16_t* Xb = (bf16_t*)(ws);             // 8 MB: X bf16 [4096][1024]
  bf16_t* Wt = (bf16_t*)(ws + 8 * MB);    // 6 MB: Wq^T,Wk^T,Wv^T bf16 [n][k]
  bf16_t* Qb = (bf16_t*)(ws + 14 * MB);   // 8 MB: Q bf16 [4096][1024]
  bf16_t* Kb = (bf16_t*)(ws + 22 * MB);   // 8 MB: K bf16 [4096][1024]
  bf16_t* Vt = (bf16_t*)(ws + 30 * MB);   // 8 MB: V bf16 transposed [b][d][s]

  k_zero<<<dim3((NB * DM + 255) / 256), 256, 0, stream>>>(out, NB * DM);
  k_cvt_x<<<dim3(ROWS * DM / 256), 256, 0, stream>>>(X, Xb);
  k_cvt_wt<<<dim3(DM * DM / 256, 3), 256, 0, stream>>>(Wq, Wk, Wv, Wt);
  k_qkv_gemm<<<dim3(512, 3), 256, 0, stream>>>(Xb, Wt, bq, bk, bv, Qb, Kb, Vt);
  k_attn<<<dim3(NB * NH * (SEQ / 16)), 32, 0, stream>>>(Qb, Kb, Vt, out);
}
